// BlockDiagonalSpatialAttention_103079215455
// MI455X (gfx1250) — matrix operand order repaired and checked by reading, not compiled, not run on hardware
//
#include <hip/hip_runtime.h>
#include <hip/hip_bf16.h>

// MI455X / gfx1250, wave32. One workgroup (128 threads = 4 waves) per 7x7
// patch group. Matmuls via V_WMMA_F32_16X16X32_BF16 (f32 accumulate).

typedef __attribute__((ext_vector_type(16))) __bf16 v16bf;
typedef __attribute__((ext_vector_type(8)))  float  v8f;

#define BB   4
#define CC   64
#define HH   448
#define PP   49      // 7*7
#define HN   64      // 448/7
#define NGRP 16384   // BB * HN * HN
#define SBS  72      // bf16 LDS row stride (flat / L)
#define GS   65      // f32  LDS row stride (G / Ec)

// d_out regions (flat element offsets, in reference return order)
#define OFF_OUT 0u
#define OFF_SC  51380224u
#define OFF_COV 90718208u
#define OFF_EC  130056192u

__global__ __launch_bounds__(128)
void bdsa_kernel(const float* __restrict__ x,
                 const float* __restrict__ beta,
                 float* __restrict__ out) {
  __shared__ __bf16 sB[64 * SBS];   // flat[p][c] in bf16, rows 49..63 zero
  __shared__ __bf16 sL[64 * SBS];   // L = Sc + 0.5*cov, zero padded
  __shared__ float  sG[64 * GS];    // G, later reused for Ec
  __shared__ float  sMean[64], sMax[64], sSumInv[64];

  const int g    = blockIdx.x;            // 0..16383
  const int b    = g >> 12;               // / 4096
  const int hb   = (g >> 6) & 63;
  const int wb   = g & 63;
  const int t    = threadIdx.x;           // 0..127
  const int wv   = t >> 5;                // wave id 0..3 -> tile-row strip
  const int lane = t & 31;
  const int l16  = lane & 15;
  const int lg   = lane >> 4;             // lane group (0: lanes 0-15, 1: 16-31)

  const float betav = beta[0];
  const float* xb = x + (size_t)b * CC * HH * HH;

  // ---- zero LDS padding (disjoint from fills below; one barrier covers all)
  for (int i = t; i < 15 * SBS; i += 128) sB[49 * SBS + i] = (__bf16)0.0f;
  for (int i = t; i < 64 * SBS; i += 128) sL[i]            = (__bf16)0.0f;

  // ---- load patch: sB[p][c] = x[b][c][hb*7 + p/7][wb*7 + p%7]
  for (int i = t; i < CC * PP; i += 128) {
    int c = i / PP, p = i - c * PP;
    int hh = hb * 7 + p / 7, ww = wb * 7 + p % 7;
    float v = xb[((size_t)c * HH + hh) * HH + ww];
    sB[p * SBS + c] = (__bf16)v;
  }
  __syncthreads();

  // ---- G = flat @ flat^T  (pad 49->64), wave wv owns rows [16wv,16wv+16)
  for (int n = 0; n < 4; ++n) {
    v8f acc = {};
    #pragma unroll
    for (int k0 = 0; k0 < 64; k0 += 32) {
      v16bf a, bmat;
      #pragma unroll
      for (int v = 0; v < 8; ++v) {
        #pragma unroll
        for (int h = 0; h < 2; ++h) {
          // A (16x32 bf16): K = [0..7|8..15] per lane group, +16 for v>=4
          int ka = k0 + ((v < 4) ? 0 : 16) + lg * 8 + 2 * (v & 3) + h;
          a[2 * v + h] = sB[(16 * wv + l16) * SBS + ka];
          // B (32x16 bf16): K = [0..7|8..15] per lane group, +16 for v>=4
          int kb = k0 + ((v < 4) ? 0 : 16) + lg * 8 + 2 * (v & 3) + h;
          // B[k=c][nq] = flat[q][c]  -> gram against transposed flat
          bmat[2 * v + h] = sB[(16 * n + l16) * SBS + kb];
        }
      }
      acc = __builtin_amdgcn_wmma_f32_16x16x32_bf16(false, a, false, bmat,
                                                    (short)0, acc, false, false);
    }
    #pragma unroll
    for (int r = 0; r < 8; ++r)     // C/D: M = r + 8*lg, N = l16
      sG[(16 * wv + r + lg * 8) * GS + 16 * n + l16] = acc[r];
  }
  __syncthreads();

  // ---- per-row stats: channel mean, row max, 1/sum(exp)
  if (t < PP) {
    float s = 0.f;
    for (int c = 0; c < CC; ++c) s += (float)sB[t * SBS + c];
    sMean[t] = s * (1.0f / CC);
    float mx = -3.4e38f;
    for (int q = 0; q < PP; ++q) mx = fmaxf(mx, sG[t * GS + q]);
    float es = 0.f;
    for (int q = 0; q < PP; ++q) es += __expf(sG[t * GS + q] - mx);
    sMax[t]    = mx;
    sSumInv[t] = 1.0f / es;
  }
  __syncthreads();

  // ---- Sc & cov (contiguous per group -> coalesced NT stores), build L(bf16)
  {
    float* scO  = out + OFF_SC  + (size_t)g * (PP * PP);
    float* covO = out + OFF_COV + (size_t)g * (PP * PP);
    for (int i = t; i < PP * PP; i += 128) {
      int p = i / PP, q = i - p * PP;
      float gv = sG[p * GS + q];
      float sc = __expf(gv - sMax[p]) * sSumInv[p];
      float cv = gv * (1.0f / CC) - sMean[p] * sMean[q];   // cov = G/C - m m^T
      __builtin_nontemporal_store(sc, scO + i);
      __builtin_nontemporal_store(cv, covO + i);
      sL[p * SBS + q] = (__bf16)(sc + 0.5f * cv);          // ALPHA = 0.5
    }
  }
  __syncthreads();

  // ---- Ec = L @ flat  (K = padded q dim = 64), result overwrites sG
  for (int n = 0; n < 4; ++n) {
    v8f acc = {};
    #pragma unroll
    for (int k0 = 0; k0 < 64; k0 += 32) {
      v16bf a, bmat;
      #pragma unroll
      for (int v = 0; v < 8; ++v) {
        #pragma unroll
        for (int h = 0; h < 2; ++h) {
          int ka = k0 + ((v < 4) ? 0 : 16) + lg * 8 + 2 * (v & 3) + h;
          a[2 * v + h] = sL[(16 * wv + l16) * SBS + ka];
          int kb = k0 + ((v < 4) ? 0 : 16) + lg * 8 + 2 * (v & 3) + h;
          // B[k=q][nc] = flat[q][c]
          bmat[2 * v + h] = sB[kb * SBS + 16 * n + l16];
        }
      }
      acc = __builtin_amdgcn_wmma_f32_16x16x32_bf16(false, a, false, bmat,
                                                    (short)0, acc, false, false);
    }
    #pragma unroll
    for (int r = 0; r < 8; ++r)
      sG[(16 * wv + r + lg * 8) * GS + 16 * n + l16] = acc[r];  // Ec[p][c]
  }
  __syncthreads();

  // ---- fold + fused elementwise:  out = x*(beta*Ec + x),  Ec_map = Ec
  {
    float* outO = out + OFF_OUT;
    float* ecO  = out + OFF_EC;
    for (int i = t; i < CC * PP; i += 128) {
      int c = i / PP, p = i - c * PP;
      int hh = hb * 7 + p / 7, ww = wb * 7 + p % 7;
      size_t gi = ((size_t)(b * CC + c) * HH + hh) * HH + ww;
      float ec = sG[p * GS + c];
      float xv = x[gi];
      float att = betav * ec + xv;
      __builtin_nontemporal_store(xv * att, outO + gi);
      __builtin_nontemporal_store(ec, ecO + gi);
    }
  }
}

extern "C" void kernel_launch(void* const* d_in, const int* in_sizes, int n_in,
                              void* d_out, int out_size, void* d_ws, size_t ws_size,
                              hipStream_t stream) {
  const float* x    = (const float*)d_in[0];
  const float* beta = (const float*)d_in[1];
  float* out        = (float*)d_out;
  (void)in_sizes; (void)n_in; (void)out_size; (void)d_ws; (void)ws_size;
  hipLaunchKernelGGL(bdsa_kernel, dim3(NGRP), dim3(128), 0, stream, x, beta, out);
}
